// MultiHeadAttention_63127429317150
// MI455X (gfx1250) — matrix operand order repaired and checked by reading, not compiled, not run on hardware
//
#include <hip/hip_runtime.h>

typedef __attribute__((ext_vector_type(16))) _Float16 v16h;
typedef __attribute__((ext_vector_type(8)))  _Float16 v8h;
typedef __attribute__((ext_vector_type(8)))  float    v8f;

union V16 { v16h v; v8h h[2]; };

#define BATCH   8
#define SEQ     1024
#define DIM     768
#define NHEADS  12
#define HDIM    64
#define MROWS   (BATCH * SEQ)        // 8192
#define CTOT    (3 * DIM)            // 2304 columns (Q|K|V)
#define SCALE   0.125f               // 1/sqrt(64)

#define WMMA(A, B, C) __builtin_amdgcn_wmma_f32_16x16x32_f16( \
    false, (A), false, (B), (short)0, (C), false, false)

// ---------------------------------------------------------------------------
// Kernel 1: convert x -> f16 (row major) and W{q,k,v} -> f16 transposed
//           WT[c][k] = W_mat[k][c%768], c in [0,2304)
// ---------------------------------------------------------------------------
__global__ __launch_bounds__(256) void cvt_kernel(
    const float* __restrict__ x,
    const float* __restrict__ Wq, const float* __restrict__ Wk,
    const float* __restrict__ Wv,
    _Float16* __restrict__ Xh, _Float16* __restrict__ WT)
{
    int tid = blockIdx.x * blockDim.x + threadIdx.x;
    const int totalX = MROWS * DIM;
    if (tid < totalX) {
        Xh[tid] = (_Float16)x[tid];
    } else {
        int t = tid - totalX;
        if (t < CTOT * DIM) {
            int c = t / DIM;
            int k = t - c * DIM;
            int mat = c / DIM;
            int col = c - mat * DIM;
            const float* W = (mat == 0) ? Wq : (mat == 1 ? Wk : Wv);
            WT[(size_t)c * DIM + k] = (_Float16)W[(size_t)k * DIM + col];
        }
    }
}

// ---------------------------------------------------------------------------
// Kernel 2: QKV projection GEMM. One wave per 16x64 output tile.
//   All 10 b128 loads of a k-chunk are issued into distinct registers
//   BEFORE the 4 WMMAs, so unrolled iterations overlap loads with math.
//   Q,K stored [b,h,n,d] f16 ; V stored transposed [b,h,d,n] f16
// ---------------------------------------------------------------------------
__global__ __launch_bounds__(256) void qkv_gemm_kernel(
    const _Float16* __restrict__ Xh, const _Float16* __restrict__ WT,
    const float* __restrict__ bq, const float* __restrict__ bk,
    const float* __restrict__ bv,
    _Float16* __restrict__ Qh, _Float16* __restrict__ Kh,
    _Float16* __restrict__ Vt)
{
    const int wave = (blockIdx.x * blockDim.x + threadIdx.x) >> 5;
    const int lane = threadIdx.x & 31;
    const int lo = lane & 15;
    const int hi = lane >> 4;

    const int NT = CTOT / 64;            // 36 column super-tiles
    const int mt = wave / NT;            // row tile 0..511
    const int nt = wave - mt * NT;       // col super-tile 0..35
    if (mt >= MROWS / 16) return;

    const int r0 = mt * 16;
    const int c0 = nt * 64;

    const _Float16* arow = Xh + (size_t)(r0 + lo) * DIM;
    const _Float16* bcol[4];
#pragma unroll
    for (int j = 0; j < 4; ++j)
        bcol[j] = WT + (size_t)(c0 + j * 16 + lo) * DIM;

    v8f acc[4] = {};
    for (int kt = 0; kt < DIM; kt += 32) {
        // Issue ALL loads of this chunk first (distinct destinations).
        V16 A, Bf[4];
        A.h[0] = *(const v8h*)(arow + kt + hi * 8);
        A.h[1] = *(const v8h*)(arow + kt + hi * 8 + 16);
#pragma unroll
        for (int j = 0; j < 4; ++j) {
            Bf[j].h[0] = *(const v8h*)(bcol[j] + kt + hi * 8);
            Bf[j].h[1] = *(const v8h*)(bcol[j] + kt + hi * 8 + 16);
        }
        // Then the 4 WMMAs (A fragment reused 4x).
#pragma unroll
        for (int j = 0; j < 4; ++j)
            acc[j] = WMMA(A.v, Bf[j].v, acc[j]);
    }

    const int mat = c0 / DIM;            // whole super-tile in one matrix
    const int cbase = c0 - mat * DIM;
    const float* bias_ptr = (mat == 0) ? bq : (mat == 1 ? bk : bv);
    _Float16* dsth = (mat == 0) ? Qh : Kh;

#pragma unroll
    for (int j = 0; j < 4; ++j) {
        const int cc = cbase + j * 16 + lo;   // column within matrix
        const int hh = cc >> 6;               // head
        const int dd = cc & 63;               // dim within head
        const float bias = bias_ptr[cc];
#pragma unroll
        for (int i = 0; i < 8; ++i) {
            const int r = r0 + i + hi * 8;    // global row (C layout)
            const int b = r >> 10;
            const int n = r & 1023;
            const _Float16 hv = (_Float16)(acc[j][i] + bias);
            if (mat == 2)
                Vt[((size_t)((b * NHEADS + hh) * HDIM + dd)) * SEQ + n] = hv;
            else
                dsth[((size_t)((b * NHEADS + hh) * SEQ + n)) * HDIM + dd] = hv;
        }
    }
}

// ---------------------------------------------------------------------------
// Kernel 3: flash attention. One wave per (b,h, 32-query tile): two 16-row
//   tiles share every K and V B-fragment -> 1 b128 load per WMMA.
//   V loads are issued before the softmax VALU block so their latency is
//   hidden behind the exp/shuffle work.
// ---------------------------------------------------------------------------
__global__ __launch_bounds__(256) void attn_kernel(
    const _Float16* __restrict__ Qh, const _Float16* __restrict__ Kh,
    const _Float16* __restrict__ Vt, float* __restrict__ out)
{
    __shared__ __align__(16) _Float16 pbuf[8][32 * 32];   // 16 KB

    const int wslot = threadIdx.x >> 5;
    const int wave  = (blockIdx.x * blockDim.x + threadIdx.x) >> 5;
    const int lane  = threadIdx.x & 31;
    const int lo = lane & 15;
    const int hi = lane >> 4;

    const int qt = wave & 31;            // 32 query tiles of 32
    const int bh = wave >> 5;            // 0..95 (b*12+h)
    if (bh >= BATCH * NHEADS) return;
    const size_t base = (size_t)bh * SEQ * HDIM;   // same for Qh/Kh/Vt
    const int q0 = qt * 32;

    // Q A-fragments: [row tile u][d-chunk c], rows q0+u*16+lo
    V16 Aq[2][2];
#pragma unroll
    for (int u = 0; u < 2; ++u) {
        const _Float16* qrow = Qh + base + (size_t)(q0 + u * 16 + lo) * HDIM;
        Aq[u][0].h[0] = *(const v8h*)(qrow + hi * 8);
        Aq[u][0].h[1] = *(const v8h*)(qrow + hi * 8 + 16);
        Aq[u][1].h[0] = *(const v8h*)(qrow + 32 + hi * 8);
        Aq[u][1].h[1] = *(const v8h*)(qrow + 32 + hi * 8 + 16);
    }

    float mrow[2][8], lrow[2][8];
    v8f O[2][4] = {};
#pragma unroll
    for (int u = 0; u < 2; ++u)
#pragma unroll
        for (int i = 0; i < 8; ++i) { mrow[u][i] = -1e30f; lrow[u][i] = 0.0f; }

    _Float16* pw = &pbuf[wslot][0];
    const _Float16* vrow[4];
#pragma unroll
    for (int dt = 0; dt < 4; ++dt)
        vrow[dt] = Vt + base + (size_t)(dt * 16 + lo) * SEQ + hi * 8;

    for (int k0 = 0; k0 < SEQ; k0 += 32) {
        // K B-fragments: [key half v][d-chunk c], shared by both row tiles
        V16 Bk[2][2];
#pragma unroll
        for (int v = 0; v < 2; ++v) {
            const _Float16* krow =
                Kh + base + (size_t)(k0 + v * 16 + lo) * HDIM;
            Bk[v][0].h[0] = *(const v8h*)(krow + hi * 8);
            Bk[v][0].h[1] = *(const v8h*)(krow + hi * 8 + 16);
            Bk[v][1].h[0] = *(const v8h*)(krow + 32 + hi * 8);
            Bk[v][1].h[1] = *(const v8h*)(krow + 32 + hi * 8 + 16);
        }

        // S[u][v]: rows of tile u vs keys of half v (contraction over d=64)
        v8f S[2][2];
#pragma unroll
        for (int u = 0; u < 2; ++u)
#pragma unroll
            for (int v = 0; v < 2; ++v) {
                v8f z = {};
                z = WMMA(Aq[u][0].v, Bk[v][0].v, z);
                S[u][v] = WMMA(Aq[u][1].v, Bk[v][1].v, z);
            }

        // Issue V loads NOW; they complete while the softmax VALU runs.
        V16 Bv[4];
#pragma unroll
        for (int dt = 0; dt < 4; ++dt) {
            Bv[dt].h[0] = *(const v8h*)(vrow[dt] + k0);
            Bv[dt].h[1] = *(const v8h*)(vrow[dt] + k0 + 16);
        }

        // online softmax per row; rows live across the 16-lane halves
#pragma unroll
        for (int u = 0; u < 2; ++u) {
#pragma unroll
            for (int i = 0; i < 8; ++i) {
                const float a = S[u][0][i] * SCALE;
                const float c = S[u][1][i] * SCALE;
                float t = fmaxf(a, c);
                t = fmaxf(t, __shfl_xor(t, 1, 32));
                t = fmaxf(t, __shfl_xor(t, 2, 32));
                t = fmaxf(t, __shfl_xor(t, 4, 32));
                t = fmaxf(t, __shfl_xor(t, 8, 32));
                const float nm = fmaxf(mrow[u][i], t);
                const float sc = __expf(mrow[u][i] - nm);
                const float p0 = __expf(a - nm);
                const float p1 = __expf(c - nm);
                float rs = p0 + p1;
                rs += __shfl_xor(rs, 1, 32);
                rs += __shfl_xor(rs, 2, 32);
                rs += __shfl_xor(rs, 4, 32);
                rs += __shfl_xor(rs, 8, 32);
                lrow[u][i] = lrow[u][i] * sc + rs;
                mrow[u][i] = nm;
                O[u][0][i] *= sc; O[u][1][i] *= sc;
                O[u][2][i] *= sc; O[u][3][i] *= sc;
                // stash P (32 rows x 32 keys) in LDS in row-major C order
                const int row = u * 16 + i + hi * 8;
                pw[row * 32 + lo]      = (_Float16)p0;
                pw[row * 32 + 16 + lo] = (_Float16)p1;
            }
        }

        // reload P as A-fragments (16x32 each): row = u*16+lo,
        // elems hi*8+{0..7} and hi*8+16+{0..7}
        V16 Ap[2];
#pragma unroll
        for (int u = 0; u < 2; ++u) {
            Ap[u].h[0] = *(const v8h*)(pw + (u * 16 + lo) * 32 + hi * 8);
            Ap[u].h[1] = *(const v8h*)(pw + (u * 16 + lo) * 32 + hi * 8 + 16);
        }

        // O += P(16x32) * V(32x16); each Bv fragment shared by both row tiles
#pragma unroll
        for (int dt = 0; dt < 4; ++dt) {
            O[0][dt] = WMMA(Ap[0].v, Bv[dt].v, O[0][dt]);
            O[1][dt] = WMMA(Ap[1].v, Bv[dt].v, O[1][dt]);
        }
    }

    // epilogue: normalize and write [b, n, h*64 + d] fp32
    const int b = bh / NHEADS;
    const int h = bh - b * NHEADS;
#pragma unroll
    for (int u = 0; u < 2; ++u) {
#pragma unroll
        for (int i = 0; i < 8; ++i) {
            const int q = q0 + u * 16 + i + hi * 8;
            const float inv = 1.0f / lrow[u][i];
            float* op = out + ((size_t)(b * SEQ + q)) * DIM + h * HDIM + lo;
            op[0]  = O[u][0][i] * inv;
            op[16] = O[u][1][i] * inv;
            op[32] = O[u][2][i] * inv;
            op[48] = O[u][3][i] * inv;
        }
    }
}

// ---------------------------------------------------------------------------
extern "C" void kernel_launch(void* const* d_in, const int* in_sizes, int n_in,
                              void* d_out, int out_size, void* d_ws,
                              size_t ws_size, hipStream_t stream)
{
    const float* x  = (const float*)d_in[0];
    const float* Wq = (const float*)d_in[1];
    const float* bq = (const float*)d_in[2];
    const float* Wk = (const float*)d_in[3];
    const float* bk = (const float*)d_in[4];
    const float* Wv = (const float*)d_in[5];
    const float* bv = (const float*)d_in[6];
    float* out = (float*)d_out;

    // workspace layout (f16 buffers, all 256B-aligned offsets)
    char* ws = (char*)d_ws;
    const size_t XH_B  = (size_t)MROWS * DIM * 2;                 // 12.6 MB
    const size_t WT_B  = (size_t)CTOT  * DIM * 2;                 //  3.5 MB
    const size_t QKV_B = (size_t)BATCH * NHEADS * SEQ * HDIM * 2; // 12.6 MB
    _Float16* Xh = (_Float16*)(ws);
    _Float16* WT = (_Float16*)(ws + XH_B);
    _Float16* Qh = (_Float16*)(ws + XH_B + WT_B);
    _Float16* Kh = (_Float16*)(ws + XH_B + WT_B + QKV_B);
    _Float16* Vt = (_Float16*)(ws + XH_B + WT_B + 2 * QKV_B);

    // 1) convert/transpose
    {
        const int total = MROWS * DIM + CTOT * DIM;  // 8,060,928
        const int blocks = (total + 255) / 256;
        cvt_kernel<<<blocks, 256, 0, stream>>>(x, Wq, Wk, Wv, Xh, WT);
    }
    // 2) QKV projection: 512 x 36 = 18432 waves, 8 waves/block
    {
        const int waves = (MROWS / 16) * (CTOT / 64);
        qkv_gemm_kernel<<<waves / 8, 256, 0, stream>>>(
            Xh, WT, bq, bk, bv, Qh, Kh, Vt);
    }
    // 3) attention: 96 * 32 = 3072 waves, 8 waves/block
    {
        attn_kernel<<<(BATCH * NHEADS * 32) / 8, 256, 0, stream>>>(
            Qh, Kh, Vt, out);
    }
}
